// FEGIN_60378650247272
// MI455X (gfx1250) — matrix operand order repaired and checked by reading, not compiled, not run on hardware
//
#include <hip/hip_runtime.h>

// ---------------------------------------------------------------------------
// Problem constants (match reference)
// ---------------------------------------------------------------------------
#define N_NODES 100000
#define N_EDGES 1600000
#define HID     128
#define NGRAPH  128
#define LAYERS  3
#define NCLS    4
#define BN_EPS  1e-5f

// ---------------------------------------------------------------------------
// WMMA types / helpers (CDNA5 gfx1250, wave32)
// ---------------------------------------------------------------------------
typedef __attribute__((ext_vector_type(16))) __bf16 v16bf;
typedef __attribute__((ext_vector_type(8)))  float  v8f;

union Frag {
    v16bf v;
    uint4 q[2];
    unsigned short u[16];
};

__device__ __forceinline__ unsigned short f2bf(float f) {
    // round-to-nearest-even f32 -> bf16
    unsigned u = __float_as_uint(f);
    unsigned r = u + 0x7FFFu + ((u >> 16) & 1u);
    return (unsigned short)(r >> 16);
}

// ---------------------------------------------------------------------------
// Weight swizzle: W[K,Nout] f32 -> fragment-order bf16
//   out[((nt*KB + kb)*32 + lane)*16 + j] = bf16(W[(kb*32 + (lane>>4)*16 + j)*Nout
//                                                + nt*16 + (lane&15)])
// so each lane's 16 B-fragment elements are one contiguous 32B run.
// ---------------------------------------------------------------------------
__global__ void swizzle_w(const float* __restrict__ W,
                          unsigned short* __restrict__ out,
                          int Nout, int KB)
{
    const int t = blockIdx.x * blockDim.x + threadIdx.x;
    const int total = (Nout / 16) * KB * 512;
    if (t >= total) return;
    const int j    = t & 15;
    const int lane = (t >> 4) & 31;
    const int rest = t >> 9;
    const int kbi  = rest % KB;
    const int nt   = rest / KB;
    const int hi   = lane >> 4;
    const int n    = nt * 16 + (lane & 15);
    const int k    = kbi * 32 + hi * 16 + j;
    out[t] = f2bf(W[(size_t)k * Nout + n]);
}

// ---------------------------------------------------------------------------
// Embedding gather -> bf16 row-major xcat [N, 384]
// ---------------------------------------------------------------------------
__global__ void embed_gather(const int* __restrict__ xidx,
                             const float* __restrict__ node_emb,
                             const float* __restrict__ comp_emb,
                             const float* __restrict__ pin_emb,
                             unsigned short* __restrict__ xcat)
{
    const long long t = (long long)blockIdx.x * blockDim.x + threadIdx.x;
    if (t >= (long long)N_NODES * 3 * HID) return;
    const int n   = (int)(t / (3 * HID));
    const int j   = (int)(t % (3 * HID));
    const int tbl = j >> 7;
    const int c   = j & 127;
    const float* tab = (tbl == 0) ? node_emb : (tbl == 1) ? comp_emb : pin_emb;
    const int idx = max(xidx[n * 3 + tbl], 0);
    xcat[t] = f2bf(tab[(size_t)idx * HID + c]);
}

// ---------------------------------------------------------------------------
// f32 -> bf16 convert (row-major, same shape)
// ---------------------------------------------------------------------------
__global__ void cvt_bf16(const float* __restrict__ src,
                         unsigned short* __restrict__ dst, int n)
{
    const int t = blockIdx.x * blockDim.x + threadIdx.x;
    if (t < n) dst[t] = f2bf(src[t]);
}

// ---------------------------------------------------------------------------
// Generic WMMA GEMM:  Y[M,Nout] = act(Abf[M,K] @ W + bias)
//   Abf : bf16 row-major (K multiple of 32); A row index clamped to M-1, so
//         all A loads are unconditional (out-of-range rows compute duplicate
//         results that the guarded store drops).
//   Wsw : bf16 pre-swizzled fragment order (see swizzle_w)
// Template K -> fully unrolled loop with immediate-offset b128 loads.
// Block: 256 threads = 8 waves -> 16 rows x 128 cols; grid.y extends columns.
// ---------------------------------------------------------------------------
template<int K, bool OUTBF, bool RELU>
__global__ void gemm_wmma(const unsigned short* __restrict__ Abf,
                          const unsigned short* __restrict__ Wsw,
                          const float* __restrict__ bias,
                          float* __restrict__ Yf,
                          unsigned short* __restrict__ Ybf,
                          int M, int Nout)
{
    constexpr int KB = K / 32;
    const int lane = threadIdx.x & 31;
    const int wave = threadIdx.x >> 5;
    const int m    = lane & 15;
    const int hi   = lane >> 4;
    const int gm   = min(blockIdx.x * 16 + m, M - 1);   // clamp, not mask
    const int nt   = blockIdx.y * 8 + wave;
    const int n    = nt * 16 + (lane & 15);

    const uint4* aq = (const uint4*)(Abf + (size_t)gm * K) + hi * 2;  // 8 bf16/uint4
    const uint4* bq = (const uint4*)Wsw + ((size_t)nt * KB * 32 + lane) * 2;

    v8f acc = {};
#pragma unroll
    for (int kb = 0; kb < KB; ++kb) {
        Frag a, b;
        // A: lane holds K = kb*32 + hi*16 .. +15 (contiguous)
        a.q[0] = aq[kb * 4];
        a.q[1] = aq[kb * 4 + 1];
        // B: contiguous 32B per lane in swizzled weights
        b.q[0] = bq[kb * 64];
        b.q[1] = bq[kb * 64 + 1];
        acc = __builtin_amdgcn_wmma_f32_16x16x32_bf16(false, a.v, false, b.v,
                                                      (short)0, acc, false, false);
    }

    const float bv = bias[n];
#pragma unroll
    for (int v = 0; v < 8; ++v) {                 // D: VGPR v -> row hi*8 + v
        const int row = blockIdx.x * 16 + hi * 8 + v;
        if (row < M) {
            float val = acc[v] + bv;
            if (RELU) val = fmaxf(val, 0.f);
            if (OUTBF) Ybf[(size_t)row * Nout + n] = f2bf(val);
            else       Yf [(size_t)row * Nout + n] = val;
        }
    }
}

// ---------------------------------------------------------------------------
// Edge aggregation  z[dst] += h[src]   (z pre-seeded with h)
// ---------------------------------------------------------------------------
__global__ void edge_agg(const int* __restrict__ ei,
                         const float* __restrict__ h,
                         float* __restrict__ z)
{
    const long long t = (long long)blockIdx.x * blockDim.x + threadIdx.x;
    if (t >= (long long)N_EDGES * 32) return;
    const int e  = (int)(t >> 5);
    const int cg = ((int)t & 31) * 4;
    const int s  = ei[e];
    const int d  = ei[N_EDGES + e];
    const float4 v = *(const float4*)(h + (size_t)s * HID + cg);
    float* zp = z + (size_t)d * HID + cg;
    unsafeAtomicAdd(zp + 0, v.x);
    unsafeAtomicAdd(zp + 1, v.y);
    unsafeAtomicAdd(zp + 2, v.z);
    unsafeAtomicAdd(zp + 3, v.w);
}

// ---------------------------------------------------------------------------
// BatchNorm (training-mode batch statistics)
// ---------------------------------------------------------------------------
__global__ void bn_stats(const float* __restrict__ z, float* __restrict__ sums)
{
    const int t = blockIdx.x * blockDim.x + threadIdx.x;
    const int c = t & (HID - 1);
    const int r0 = t >> 7;
    const int stride = (gridDim.x * blockDim.x) >> 7;
    float s = 0.f, q = 0.f;
    for (int r = r0; r < N_NODES; r += stride) {
        const float v = z[(size_t)r * HID + c];
        s += v;
        q += v * v;
    }
    unsafeAtomicAdd(sums + c, s);
    unsafeAtomicAdd(sums + HID + c, q);
}

__global__ void bn_final(const float* __restrict__ sums,
                         const float* __restrict__ gamma,
                         const float* __restrict__ beta,
                         float* __restrict__ sc)
{
    const int c = threadIdx.x;
    if (c < HID) {
        const float mu  = sums[c] * (1.0f / (float)N_NODES);
        const float var = sums[HID + c] * (1.0f / (float)N_NODES) - mu * mu;
        const float inv = rsqrtf(var + BN_EPS);
        const float scale = gamma[c] * inv;
        sc[c]       = scale;
        sc[HID + c] = beta[c] - mu * scale;
    }
}

__global__ void bn_apply(const float* __restrict__ z,
                         float* __restrict__ h,
                         const float* __restrict__ sc,
                         int residual)
{
    const int t = blockIdx.x * blockDim.x + threadIdx.x;
    if (t >= N_NODES * HID) return;
    const int c = t & (HID - 1);
    float v = fmaxf(z[t] * sc[c] + sc[HID + c], 0.f);
    if (residual) v += h[t];
    h[t] = v;
}

// ---------------------------------------------------------------------------
// Pooling: per-graph sum / max / count (h >= 0, so uint-bit atomicMax is exact;
// zero-init matches reference isfinite->0 for empty graphs)
// ---------------------------------------------------------------------------
__global__ void pool_kernel(const float* __restrict__ h,
                            const int* __restrict__ batch,
                            float* __restrict__ counts,
                            float* __restrict__ gsum,
                            unsigned* __restrict__ gmax)
{
    const long long t = (long long)blockIdx.x * blockDim.x + threadIdx.x;
    if (t >= (long long)N_NODES * 32) return;
    const int nrow = (int)(t >> 5);
    const int cg   = ((int)t & 31) * 4;
    const int g    = batch[nrow];
    const float4 v = *(const float4*)(h + (size_t)nrow * HID + cg);
    float* sp = gsum + (size_t)g * HID + cg;
    unsafeAtomicAdd(sp + 0, v.x);
    unsafeAtomicAdd(sp + 1, v.y);
    unsafeAtomicAdd(sp + 2, v.z);
    unsafeAtomicAdd(sp + 3, v.w);
    unsigned* mp = gmax + (size_t)g * HID + cg;
    atomicMax(mp + 0, __float_as_uint(v.x));
    atomicMax(mp + 1, __float_as_uint(v.y));
    atomicMax(mp + 2, __float_as_uint(v.z));
    atomicMax(mp + 3, __float_as_uint(v.w));
    if (cg == 0) unsafeAtomicAdd(counts + g, 1.0f);
}

__global__ void emb_build(const float* __restrict__ counts,
                          const float* __restrict__ gsum,
                          const unsigned* __restrict__ gmax,
                          unsigned short* __restrict__ embbf)  // [G, 3H] bf16
{
    const int t = blockIdx.x * blockDim.x + threadIdx.x;
    if (t >= NGRAPH * 3 * HID) return;
    const int g   = t / (3 * HID);
    const int j   = t % (3 * HID);
    const int sec = j / HID;
    const int c   = j % HID;
    const float s = gsum[g * HID + c];
    float out;
    if (sec == 0)      out = s / fmaxf(counts[g], 1.0f);
    else if (sec == 1) out = __uint_as_float(gmax[g * HID + c]);
    else               out = s;
    embbf[t] = f2bf(out);
}

// ---------------------------------------------------------------------------
// Classifier head: [G,128] @ [128,4] + b
// ---------------------------------------------------------------------------
__global__ void classifier(const float* __restrict__ f2,
                           const float* __restrict__ W,
                           const float* __restrict__ b,
                           float* __restrict__ out)
{
    const int t = blockIdx.x * blockDim.x + threadIdx.x;
    if (t >= NGRAPH * NCLS) return;
    const int g = t >> 2;
    const int c = t & 3;
    float s = b[c];
#pragma unroll 8
    for (int k = 0; k < HID; ++k)
        s += f2[g * HID + k] * W[k * NCLS + c];
    out[t] = s;
}

__global__ void zero_kernel(float* __restrict__ p, int n)
{
    const int t = blockIdx.x * blockDim.x + threadIdx.x;
    if (t < n) p[t] = 0.f;
}

// ---------------------------------------------------------------------------
// Launch
// ---------------------------------------------------------------------------
extern "C" void kernel_launch(void* const* d_in, const int* in_sizes, int n_in,
                              void* d_out, int out_size, void* d_ws, size_t ws_size,
                              hipStream_t stream)
{
    (void)in_sizes; (void)n_in; (void)out_size; (void)ws_size;

    const int*   x_idx    = (const int*)  d_in[0];
    const int*   edge_idx = (const int*)  d_in[1];
    const int*   batch    = (const int*)  d_in[2];
    const float* node_emb = (const float*)d_in[3];
    const float* comp_emb = (const float*)d_in[4];
    const float* pin_emb  = (const float*)d_in[5];
    const float* proj_w   = (const float*)d_in[6];
    const float* proj_b   = (const float*)d_in[7];
    const float* gin_w1   = (const float*)d_in[8];
    const float* gin_b1   = (const float*)d_in[9];
    const float* gin_w2   = (const float*)d_in[10];
    const float* gin_b2   = (const float*)d_in[11];
    const float* bn_gamma = (const float*)d_in[12];
    const float* bn_beta  = (const float*)d_in[13];
    const float* fus_w1   = (const float*)d_in[14];
    const float* fus_b1   = (const float*)d_in[15];
    const float* fus_w2   = (const float*)d_in[16];
    const float* fus_b2   = (const float*)d_in[17];
    const float* cls_w    = (const float*)d_in[18];
    const float* cls_b    = (const float*)d_in[19];
    float* out = (float*)d_out;

    // ---- workspace layout (16B-aligned cursor) --------------------------
    const size_t NH = (size_t)N_NODES * HID;
    char* cur = (char*)d_ws;
    auto take = [&](size_t bytes) {
        char* p = cur;
        cur += (bytes + 15) & ~(size_t)15;
        return (void*)p;
    };
    float*          h      = (float*)take(NH * 4);                       // [N,128]
    float*          z      = (float*)take(NH * 4);                       // [N,128]
    unsigned short* zbf    = (unsigned short*)take(NH * 2);              // [N,128]
    unsigned short* xcat   = (unsigned short*)take((size_t)N_NODES*384*2);
    unsigned short* ybf    = xcat;   // overlays xcat (dead after projection)
    unsigned short* w_proj = (unsigned short*)take(49152 * 2);
    unsigned short* w_g1   = (unsigned short*)take(3 * 32768 * 2);
    unsigned short* w_g2   = (unsigned short*)take(3 * 32768 * 2);
    unsigned short* w_f1   = (unsigned short*)take(98304 * 2);
    unsigned short* w_f2   = (unsigned short*)take(32768 * 2);
    float*          bnsums = (float*)take(2 * HID * 4);
    float*          bnsc   = (float*)take(2 * HID * 4);
    float*          counts = (float*)take(NGRAPH * 4);
    float*          gsum   = (float*)take((size_t)NGRAPH * HID * 4);
    unsigned*       gmax   = (unsigned*)take((size_t)NGRAPH * HID * 4);
    unsigned short* embbf  = (unsigned short*)take((size_t)NGRAPH * 3 * HID * 2);
    unsigned short* f1bf   = (unsigned short*)take((size_t)NGRAPH * 2 * HID * 2);
    float*          f2     = (float*)take((size_t)NGRAPH * HID * 4);

    const int mtiles = (N_NODES + 15) / 16;          // 6250

    // ---- weight pre-swizzle (once per launch, ~0.75 MB total) -----------
    swizzle_w<<<(49152 + 255) / 256, 256, 0, stream>>>(proj_w, w_proj, HID, 12);
    for (int l = 0; l < LAYERS; ++l) {
        swizzle_w<<<(32768 + 255) / 256, 256, 0, stream>>>(
            gin_w1 + (size_t)l * HID * 2 * HID, w_g1 + (size_t)l * 32768, 2 * HID, 4);
        swizzle_w<<<(32768 + 255) / 256, 256, 0, stream>>>(
            gin_w2 + (size_t)l * 2 * HID * HID, w_g2 + (size_t)l * 32768, HID, 8);
    }
    swizzle_w<<<(98304 + 255) / 256, 256, 0, stream>>>(fus_w1, w_f1, 2 * HID, 12);
    swizzle_w<<<(32768 + 255) / 256, 256, 0, stream>>>(fus_w2, w_f2, HID, 8);

    // ---- embedding gather + projection -> h -----------------------------
    {
        const long long work = (long long)N_NODES * 3 * HID;
        embed_gather<<<(int)((work + 255) / 256), 256, 0, stream>>>(
            x_idx, node_emb, comp_emb, pin_emb, xcat);
        gemm_wmma<384, false, false><<<dim3(mtiles, 1), 256, 0, stream>>>(
            xcat, w_proj, proj_b, h, (unsigned short*)nullptr, N_NODES, HID);
    }

    // ---- GIN layers ------------------------------------------------------
    for (int l = 0; l < LAYERS; ++l) {
        hipMemcpyAsync(z, h, NH * sizeof(float), hipMemcpyDeviceToDevice, stream);
        {
            const long long work = (long long)N_EDGES * 32;
            edge_agg<<<(int)((work + 255) / 256), 256, 0, stream>>>(edge_idx, h, z);
        }
        cvt_bf16<<<(int)((NH + 255) / 256), 256, 0, stream>>>(z, zbf, (int)NH);
        // ybf = relu(z @ W1 + b1)   [N,256] bf16
        gemm_wmma<128, true, true><<<dim3(mtiles, 2), 256, 0, stream>>>(
            zbf, w_g1 + (size_t)l * 32768, gin_b1 + (size_t)l * 2 * HID,
            (float*)nullptr, ybf, N_NODES, 2 * HID);
        // z = ybf @ W2 + b2         [N,128] f32
        gemm_wmma<256, false, false><<<dim3(mtiles, 1), 256, 0, stream>>>(
            ybf, w_g2 + (size_t)l * 32768, gin_b2 + (size_t)l * HID,
            z, (unsigned short*)nullptr, N_NODES, HID);
        // batch-norm + relu + residual
        zero_kernel<<<1, 256, 0, stream>>>(bnsums, 2 * HID);
        bn_stats<<<1024, 256, 0, stream>>>(z, bnsums);
        bn_final<<<1, 128, 0, stream>>>(bnsums,
                                        bn_gamma + (size_t)l * HID,
                                        bn_beta + (size_t)l * HID, bnsc);
        bn_apply<<<(int)((NH + 255) / 256), 256, 0, stream>>>(z, h, bnsc, l > 0);
    }

    // ---- pooling ---------------------------------------------------------
    zero_kernel<<<(NGRAPH + 2 * NGRAPH * HID + 255) / 256, 256, 0, stream>>>(
        counts, NGRAPH + 2 * NGRAPH * HID);
    {
        const long long work = (long long)N_NODES * 32;
        pool_kernel<<<(int)((work + 255) / 256), 256, 0, stream>>>(
            h, batch, counts, gsum, gmax);
    }
    emb_build<<<(NGRAPH * 3 * HID + 255) / 256, 256, 0, stream>>>(
        counts, gsum, gmax, embbf);

    // ---- fusion head -----------------------------------------------------
    gemm_wmma<384, true, true><<<dim3(NGRAPH / 16, 2), 256, 0, stream>>>(
        embbf, w_f1, fus_b1, (float*)nullptr, f1bf, NGRAPH, 2 * HID);
    gemm_wmma<256, false, true><<<dim3(NGRAPH / 16, 1), 256, 0, stream>>>(
        f1bf, w_f2, fus_b2, f2, (unsigned short*)nullptr, NGRAPH, HID);
    classifier<<<(NGRAPH * NCLS + 255) / 256, 256, 0, stream>>>(
        f2, cls_w, cls_b, out);
}
